// MultiLatentAttention_31774168056146
// MI455X (gfx1250) — matrix operand order repaired and checked by reading, not compiled, not run on hardware
//
#include <hip/hip_runtime.h>
#include <hip/hip_bf16.h>

// ---------------- problem constants ----------------
#define B_       2
#define T_       2048
#define H_       16
#define NOPE_    128
#define ROPE_    64
#define QK_      192            // NOPE + ROPE
#define VD_      128
#define KV_LORA_ 512
#define HID_     2048
#define M_       (B_ * T_)      // 4096 rows
#define SCALE_   0.07216878364870322f   // 1/sqrt(192)

typedef __attribute__((ext_vector_type(16))) __bf16 bf16x16;
typedef __attribute__((ext_vector_type(8)))  __bf16 bf16x8;
typedef __attribute__((ext_vector_type(8)))  float  f32x8;

__device__ __forceinline__ __bf16 f2bf(float f) { return (__bf16)f; }

// A-operand pattern: lane holds K = [half*16 .. half*16+7] and [half*16+8 .. half*16+15]
__device__ __forceinline__ bf16x16 ld16_pair(const __bf16* plo, const __bf16* phi) {
  bf16x8 lo = *(const bf16x8*)plo;
  bf16x8 hi = *(const bf16x8*)phi;
  return __builtin_shufflevector(lo, hi, 0,1,2,3,4,5,6,7,8,9,10,11,12,13,14,15);
}
// B-operand pattern: lane holds 16 contiguous K values (K = half*16 .. half*16+15)
__device__ __forceinline__ bf16x16 ld16_contig(const __bf16* p) {
  return ld16_pair(p, p + 8);
}

__device__ __forceinline__ f32x8 wmma_bf16(bf16x16 a, bf16x16 b, f32x8 c) {
  return __builtin_amdgcn_wmma_f32_16x16x32_bf16(false, a, false, b, (short)0, c,
                                                 false, false);
}

// ---------------- Tensor Data Mover (CDNA5 TDM) ----------------
#if __has_builtin(__builtin_amdgcn_tensor_load_to_lds)
#define HAVE_TDM 1
typedef __attribute__((ext_vector_type(4))) unsigned int u32x4;
typedef __attribute__((ext_vector_type(8))) int          i32x8;
typedef __attribute__((ext_vector_type(4))) int          i32x4;

// 2D tile DMA: global (row-major, row stride = stride_elems, element = 2B bf16)
// -> LDS packed row-major [rows][32].  tensor_rows bounds dim1 (OOB rows read 0).
__device__ __forceinline__ void tdm_load_tile_2d(const __bf16* gtile,
                                                 unsigned lds_byte_off,
                                                 unsigned tile_rows,
                                                 unsigned stride_elems,
                                                 unsigned tensor_rows) {
  unsigned long long ga = (unsigned long long)(uintptr_t)gtile;
  u32x4 g0;
  g0[0] = 1u;                                            // count=1, user desc
  g0[1] = lds_byte_off;                                  // lds_addr
  g0[2] = (unsigned)ga;                                  // global_addr[31:0]
  g0[3] = ((unsigned)(ga >> 32) & 0x01FFFFFFu)           // global_addr[56:32]
          | 0x80000000u;                                 // type=2 ("image")
  const unsigned td0 = stride_elems;                     // tensor_dim0
  const unsigned td1 = tensor_rows;                      // tensor_dim1
  const unsigned tile0 = 32u;                            // tile_dim0 (K chunk)
  i32x8 g1;
  g1[0] = 0x00010000;                                    // data_size=1 (2 bytes)
  g1[1] = (int)((td0 & 0xFFFFu) << 16);                  // tensor_dim0[15:0]
  g1[2] = (int)((td0 >> 16) | ((td1 & 0xFFFFu) << 16));  // dim0[31:16]|dim1[15:0]
  g1[3] = (int)((td1 >> 16) | (tile0 << 16));            // dim1[31:16]|tile_dim0
  g1[4] = (int)tile_rows;                                // tile_dim1 (tile_dim2=0)
  g1[5] = (int)stride_elems;                             // tensor_dim0_stride lo32
  g1[6] = 0;                                             // stride0 hi | stride1 lo
  g1[7] = 0;                                             // stride1 hi
  i32x4 z4 = {};
  i32x8 z8 = {};
  __builtin_amdgcn_tensor_load_to_lds(g0, g1, z4, z4, z8, 0);
}
#else
#define HAVE_TDM 0
#endif

// ---------------- conversion kernels ----------------
__global__ __launch_bounds__(256) void cvt_f32_bf16(const float* __restrict__ in,
                                                    __bf16* __restrict__ out, int n) {
  int i = blockIdx.x * 256 + threadIdx.x;
  if (i < n) out[i] = f2bf(in[i]);
}

// in: [K, N] fp32 row-major  ->  out: [N, K] bf16 row-major (B^T for GEMM)
__global__ __launch_bounds__(256) void cvt_transpose_bf16(const float* __restrict__ in,
                                                          __bf16* __restrict__ out,
                                                          int K, int N) {
  int e = blockIdx.x * 256 + threadIdx.x;
  if (e < K * N) {
    int k = e / N, n = e - k * N;
    out[(size_t)n * K + k] = f2bf(in[e]);
  }
}

// ---------------- generic bf16 GEMM: C[M,N] = A[M,K] * Bt[N,K]^T ----------------
// block = 256 threads (8 waves), block tile 128x128, K step 32.
// Double-buffered LDS fed by the Tensor Data Mover (wave 0 issues the DMA,
// everyone syncs on its s_wait_tensorcnt through the barrier).
__global__ __launch_bounds__(256) void gemm_bf16_nt(const __bf16* __restrict__ A,
                                                    const __bf16* __restrict__ Bt,
                                                    float* __restrict__ C,
                                                    int Ndim, int Kdim) {
  __shared__ __bf16 As[2][128][32];   // 16 KB (TDM writes packed rows)
  __shared__ __bf16 Bs[2][128][32];   // 16 KB
  const int tid  = threadIdx.x;
  const int lane = tid & 31, wave = tid >> 5;
  const int wm = wave & 3, wn = wave >> 2;
  const int lr = lane & 15, half = lane >> 4;
  const int m0 = blockIdx.y * 128, n0 = blockIdx.x * 128;
  const int nK = Kdim >> 5;
  const unsigned bRows = (unsigned)(Ndim - n0);   // rows available in B^T tile

  f32x8 acc[2][4] = {};

#if HAVE_TDM
  if (wave == 0) {
    tdm_load_tile_2d(A + (size_t)m0 * Kdim, (unsigned)(uintptr_t)&As[0][0][0],
                     128u, (unsigned)Kdim, 128u);
    tdm_load_tile_2d(Bt + (size_t)n0 * Kdim, (unsigned)(uintptr_t)&Bs[0][0][0],
                     128u, (unsigned)Kdim, bRows);
  }
#endif

  for (int it = 0; it < nK; ++it) {
    const int buf = it & 1;
    const int k0 = it << 5;
    __syncthreads();                       // readers of buf^1 are done
#if HAVE_TDM
    (void)k0;
    if (wave == 0) {
      if (it + 1 < nK) {
        int k1 = (it + 1) << 5;
        tdm_load_tile_2d(A + (size_t)m0 * Kdim + k1,
                         (unsigned)(uintptr_t)&As[buf ^ 1][0][0],
                         128u, (unsigned)Kdim, 128u);
        tdm_load_tile_2d(Bt + (size_t)n0 * Kdim + k1,
                         (unsigned)(uintptr_t)&Bs[buf ^ 1][0][0],
                         128u, (unsigned)Kdim, bRows);
        __builtin_amdgcn_s_wait_tensorcnt((short)2);   // current buf complete
      } else {
        __builtin_amdgcn_s_wait_tensorcnt((short)0);
      }
    }
#else
    // fallback: cooperative staging through VGPRs
#pragma unroll
    for (int st = 0; st < 2; ++st) {
      int c = tid + st * 256;              // 512 chunks of 8 bf16
      int r = c >> 2, cg = (c & 3) << 3;
      *(bf16x8*)&As[buf][r][cg] =
          *(const bf16x8*)(A + (size_t)(m0 + r) * Kdim + k0 + cg);
      bf16x8 bv = {};
      if ((unsigned)r < bRows)
        bv = *(const bf16x8*)(Bt + (size_t)(n0 + r) * Kdim + k0 + cg);
      *(bf16x8*)&Bs[buf][r][cg] = bv;
    }
#endif
    __syncthreads();                       // buf is valid for everyone

    bf16x16 af[2], bfrag[4];
#pragma unroll
    for (int mt = 0; mt < 2; ++mt) {
      const __bf16* p = &As[buf][wm * 32 + mt * 16 + lr][0];
      af[mt] = ld16_pair(p + half * 16, p + half * 16 + 8);
    }
#pragma unroll
    for (int nt = 0; nt < 4; ++nt)
      bfrag[nt] = ld16_contig(&Bs[buf][wn * 64 + nt * 16 + lr][half * 16]);
#pragma unroll
    for (int mt = 0; mt < 2; ++mt)
#pragma unroll
      for (int nt = 0; nt < 4; ++nt)
        acc[mt][nt] = wmma_bf16(af[mt], bfrag[nt], acc[mt][nt]);
  }

#pragma unroll
  for (int mt = 0; mt < 2; ++mt)
#pragma unroll
    for (int nt = 0; nt < 4; ++nt)
#pragma unroll
      for (int i = 0; i < 8; ++i) {
        int row = m0 + wm * 32 + mt * 16 + half * 8 + i;
        int col = n0 + wn * 64 + nt * 16 + lr;
        if (col < Ndim) C[(size_t)row * Ndim + col] = acc[mt][nt][i];
      }
}

// ---------------- RoPE on q, emit bf16 [M, H, QK] ----------------
__global__ __launch_bounds__(128) void rope_q_kernel(const float* __restrict__ q,
                                                     const float* __restrict__ freqs,
                                                     __bf16* __restrict__ qo) {
  int row = blockIdx.x / H_, h = blockIdx.x % H_;
  int tpos = row % T_;
  const float* src = q + ((size_t)row * (H_ * QK_) + (size_t)h * QK_);
  __bf16* dst = qo + ((size_t)row * H_ + h) * QK_;
  int tid = threadIdx.x;
  dst[tid] = f2bf(src[tid]);                       // 128 nope elements
  if (tid < ROPE_ / 2) {
    float f = freqs[(size_t)tpos * (ROPE_ / 2) + tid];
    float c = __cosf(f), s = __sinf(f);
    float x1 = src[NOPE_ + 2 * tid], x2 = src[NOPE_ + 2 * tid + 1];
    dst[NOPE_ + 2 * tid]     = f2bf(x1 * c - x2 * s);
    dst[NOPE_ + 2 * tid + 1] = f2bf(x1 * s + x2 * c);
  }
}

// ---------------- LayerNorm(kv_lat) + RoPE(k_rope) ----------------
__global__ __launch_bounds__(256) void kv_norm_rope_kernel(
    const float* __restrict__ kv, const float* __restrict__ freqs,
    const float* __restrict__ gamma, const float* __restrict__ beta,
    __bf16* __restrict__ kvlat, __bf16* __restrict__ krope) {
  __shared__ float sdata[256];
  int row = blockIdx.x, tid = threadIdx.x;
  const float* x = kv + (size_t)row * (KV_LORA_ + ROPE_);
  float a = x[tid], b = x[tid + 256];

  sdata[tid] = a + b; __syncthreads();
  for (int ofs = 128; ofs > 0; ofs >>= 1) {
    if (tid < ofs) sdata[tid] += sdata[tid + ofs];
    __syncthreads();
  }
  float mu = sdata[0] / (float)KV_LORA_; __syncthreads();

  sdata[tid] = a * a + b * b; __syncthreads();
  for (int ofs = 128; ofs > 0; ofs >>= 1) {
    if (tid < ofs) sdata[tid] += sdata[tid + ofs];
    __syncthreads();
  }
  float var = sdata[0] / (float)KV_LORA_ - mu * mu;
  float rstd = __frsqrt_rn(var + 1e-5f);

  __bf16* out = kvlat + (size_t)row * KV_LORA_;
  out[tid]       = f2bf((a - mu) * rstd * gamma[tid] + beta[tid]);
  out[tid + 256] = f2bf((b - mu) * rstd * gamma[tid + 256] + beta[tid + 256]);

  if (tid < ROPE_ / 2) {
    int tpos = row % T_;
    float f = freqs[(size_t)tpos * (ROPE_ / 2) + tid];
    float c = __cosf(f), s = __sinf(f);
    float x1 = x[KV_LORA_ + 2 * tid], x2 = x[KV_LORA_ + 2 * tid + 1];
    krope[(size_t)row * ROPE_ + 2 * tid]     = f2bf(x1 * c - x2 * s);
    krope[(size_t)row * ROPE_ + 2 * tid + 1] = f2bf(x1 * s + x2 * c);
  }
}

// ---------------- split kvb into K (nope||rope) and V, bf16 ----------------
__global__ __launch_bounds__(256) void split_kv_kernel(const float* __restrict__ kvb,
                                                       const __bf16* __restrict__ krope,
                                                       __bf16* __restrict__ Kc,
                                                       __bf16* __restrict__ Vc) {
  int row = blockIdx.x / H_, h = blockIdx.x % H_;
  int tid = threadIdx.x;
  const float* src = kvb + (size_t)row * (H_ * (NOPE_ + VD_)) + (size_t)h * (NOPE_ + VD_);
  __bf16* kdst = Kc + ((size_t)row * H_ + h) * QK_;
  __bf16* vdst = Vc + ((size_t)row * H_ + h) * VD_;
  if (tid < NOPE_) kdst[tid] = f2bf(src[tid]);
  else             vdst[tid - NOPE_] = f2bf(src[tid]);
  if (tid < ROPE_) kdst[NOPE_ + tid] = krope[(size_t)row * ROPE_ + tid];
}

// ---------------- flash attention (causal), WMMA for QK^T and PV ----------------
// block = 128 threads (4 waves); each wave owns 16 query rows; 64 q rows / block.
// stream 32-wide K/V blocks.
__global__ __launch_bounds__(128) void mla_attn_kernel(const __bf16* __restrict__ Q,
                                                       const __bf16* __restrict__ K,
                                                       const __bf16* __restrict__ V,
                                                       __bf16* __restrict__ O) {
  __shared__ __bf16 Ks[32][QK_];       // 12 KB, row-major: B-operand contiguous
  __shared__ __bf16 Vt[VD_][32];       // 8 KB, transposed V
  __shared__ __bf16 Ps[4][16][32];     // 4 KB, per-wave P bounce
  const int tid = threadIdx.x, lane = tid & 31, wave = tid >> 5;
  const int lr = lane & 15, half = lane >> 4;
  const int nQB = T_ / 64;
  const int qb = blockIdx.x % nQB;
  const int h  = (blockIdx.x / nQB) % H_;
  const int bb = blockIdx.x / (nQB * H_);

  // Q fragments (A operand): 6 K-slices of 32
  const int qrowA = qb * 64 + wave * 16 + lr;
  const __bf16* qptr = Q + ((size_t)(bb * T_ + qrowA) * H_ + h) * QK_;
  bf16x16 qf[6];
#pragma unroll
  for (int kf = 0; kf < 6; ++kf)
    qf[kf] = ld16_pair(qptr + kf * 32 + half * 16, qptr + kf * 32 + half * 16 + 8);

  f32x8 o[8] = {};
  float rm[8], rl[8];
#pragma unroll
  for (int i = 0; i < 8; ++i) { rm[i] = -3.0e38f; rl[i] = 0.0f; }

  const int nSB = qb * 2 + 2;          // causal: s blocks up to the diagonal
  for (int sb = 0; sb < nSB; ++sb) {
    const int s0 = sb * 32;
    __syncthreads();
    // stage K block: 32 x 192 bf16
#pragma unroll
    for (int it = 0; it < 6; ++it) {
      int c = tid + it * 128;
      int r = c / 24, cg = (c % 24) * 8;
      *(bf16x8*)&Ks[r][cg] =
          *(const bf16x8*)(K + ((size_t)(bb * T_ + s0 + r) * H_ + h) * QK_ + cg);
    }
    // stage V block transposed: Vt[d][r]
#pragma unroll
    for (int it = 0; it < 32; ++it) {
      int e = tid + it * 128;
      int r = e >> 7, d = e & 127;
      Vt[d][r] = V[((size_t)(bb * T_ + s0 + r) * H_ + h) * VD_ + d];
    }
    __syncthreads();

    // scores S = Q (16x192) * K^T (192x32): two 16-col tiles
    f32x8 sa[2] = {};
#pragma unroll
    for (int j = 0; j < 2; ++j)
#pragma unroll
      for (int kf = 0; kf < 6; ++kf) {
        bf16x16 kfr = ld16_contig(&Ks[j * 16 + lr][kf * 32 + half * 16]);
        sa[j] = wmma_bf16(qf[kf], kfr, sa[j]);
      }

    // online softmax; C/D layout: element i -> row (half*8+i), col lr
    const int qglob = qb * 64 + wave * 16 + half * 8;
#pragma unroll
    for (int i = 0; i < 8; ++i) {
      int row = qglob + i;
      float p0 = 0.0f, p1 = 0.0f;
      if (row >= s0) {                 // row has at least one unmasked column
        float v0 = sa[0][i] * SCALE_, v1 = sa[1][i] * SCALE_;
        if (s0 + lr      > row) v0 = -1.0e30f;
        if (s0 + 16 + lr > row) v1 = -1.0e30f;
        float bm = fmaxf(v0, v1);
#pragma unroll
        for (int msk = 1; msk < 16; msk <<= 1) bm = fmaxf(bm, __shfl_xor(bm, msk, 32));
        float mn = fmaxf(rm[i], bm);
        float alpha = __expf(rm[i] - mn);
        rm[i] = mn;
        p0 = __expf(v0 - mn); p1 = __expf(v1 - mn);
        float ps = p0 + p1;
#pragma unroll
        for (int msk = 1; msk < 16; msk <<= 1) ps += __shfl_xor(ps, msk, 32);
        rl[i] = rl[i] * alpha + ps;
#pragma unroll
        for (int nt = 0; nt < 8; ++nt) o[nt][i] *= alpha;
      }
      Ps[wave][half * 8 + i][lr]      = f2bf(p0);
      Ps[wave][half * 8 + i][16 + lr] = f2bf(p1);
    }

    // P (16x32, A operand) * V (32x128): 8 output tiles
    const __bf16* pp = &Ps[wave][lr][0];
    bf16x16 pa = ld16_pair(pp + half * 16, pp + half * 16 + 8);
#pragma unroll
    for (int nt = 0; nt < 8; ++nt) {
      bf16x16 bv = ld16_contig(&Vt[nt * 16 + lr][half * 16]);
      o[nt] = wmma_bf16(pa, bv, o[nt]);
    }
  }

  // normalize and store attn output bf16: [M, H*VD]
#pragma unroll
  for (int nt = 0; nt < 8; ++nt)
#pragma unroll
    for (int i = 0; i < 8; ++i) {
      int row = qb * 64 + wave * 16 + half * 8 + i;
      float val = o[nt][i] / rl[i];
      O[(size_t)(bb * T_ + row) * (H_ * VD_) + h * VD_ + nt * 16 + lr] = f2bf(val);
    }
}

// ---------------- host side ----------------
extern "C" void kernel_launch(void* const* d_in, const int* in_sizes, int n_in,
                              void* d_out, int out_size, void* d_ws, size_t ws_size,
                              hipStream_t stream) {
  (void)in_sizes; (void)n_in; (void)out_size; (void)ws_size;
  const float* hidden = (const float*)d_in[0];
  const float* freqs  = (const float*)d_in[1];
  // d_in[2] = attention_mask (causal; implemented analytically)
  const float* Wq     = (const float*)d_in[3];
  const float* Wkv_a  = (const float*)d_in[4];
  const float* kv_g   = (const float*)d_in[5];
  const float* kv_b   = (const float*)d_in[6];
  const float* Wkv_b  = (const float*)d_in[7];
  const float* Wout   = (const float*)d_in[8];
  float* out = (float*)d_out;

  char* w = (char*)d_ws;
  size_t off = 0;
  auto take = [&](size_t bytes) -> char* {
    char* p = w + off;
    off += (bytes + 255) & ~(size_t)255;
    return p;
  };

  // overlap region: hs16 + q_f32 + kv_f32 (all dead before GEMM3) hosts kvb_f32
  __bf16* hs16   = (__bf16*)take((size_t)M_ * HID_ * 2);            // 16 MB
  float*  q_f32  = (float*) take((size_t)M_ * H_ * QK_ * 4);        // 48 MB
  float*  kv_f32 = (float*) take((size_t)M_ * (KV_LORA_ + ROPE_) * 4); // 9 MB
  float*  kvb_f32 = (float*)hs16;  // 64 MB fits inside the 73.7 MB region above

  __bf16* Wq_t   = (__bf16*)take((size_t)(H_ * QK_) * HID_ * 2);
  __bf16* Wkva_t = (__bf16*)take((size_t)(KV_LORA_ + ROPE_) * HID_ * 2);
  __bf16* Wkvb_t = (__bf16*)take((size_t)(H_ * (NOPE_ + VD_)) * KV_LORA_ * 2);
  __bf16* Wout_t = (__bf16*)take((size_t)HID_ * (H_ * VD_) * 2);
  __bf16* q16    = (__bf16*)take((size_t)M_ * H_ * QK_ * 2);
  __bf16* kvlat16= (__bf16*)take((size_t)M_ * KV_LORA_ * 2);
  __bf16* krope16= (__bf16*)take((size_t)M_ * ROPE_ * 2);
  __bf16* k16    = (__bf16*)take((size_t)M_ * H_ * QK_ * 2);
  __bf16* v16    = (__bf16*)take((size_t)M_ * H_ * VD_ * 2);
  __bf16* attn16 = (__bf16*)take((size_t)M_ * H_ * VD_ * 2);

  // 1) conversions
  {
    int n = M_ * HID_;
    cvt_f32_bf16<<<(n + 255) / 256, 256, 0, stream>>>(hidden, hs16, n);
  }
  {
    int n = HID_ * (H_ * QK_);
    cvt_transpose_bf16<<<(n + 255) / 256, 256, 0, stream>>>(Wq, Wq_t, HID_, H_ * QK_);
  }
  {
    int n = HID_ * (KV_LORA_ + ROPE_);
    cvt_transpose_bf16<<<(n + 255) / 256, 256, 0, stream>>>(Wkv_a, Wkva_t, HID_,
                                                            KV_LORA_ + ROPE_);
  }
  {
    int n = KV_LORA_ * (H_ * (NOPE_ + VD_));
    cvt_transpose_bf16<<<(n + 255) / 256, 256, 0, stream>>>(Wkv_b, Wkvb_t, KV_LORA_,
                                                            H_ * (NOPE_ + VD_));
  }
  {
    int n = (H_ * VD_) * HID_;
    cvt_transpose_bf16<<<(n + 255) / 256, 256, 0, stream>>>(Wout, Wout_t, H_ * VD_, HID_);
  }

  // 2) q = hidden @ Wq ; kv = hidden @ Wkv_a
  gemm_bf16_nt<<<dim3((H_ * QK_) / 128, M_ / 128), 256, 0, stream>>>(
      hs16, Wq_t, q_f32, H_ * QK_, HID_);
  gemm_bf16_nt<<<dim3((KV_LORA_ + ROPE_ + 127) / 128, M_ / 128), 256, 0, stream>>>(
      hs16, Wkva_t, kv_f32, KV_LORA_ + ROPE_, HID_);

  // 3) RoPE(q) -> bf16 ; LN(kv_lat)+RoPE(k_rope) -> bf16
  rope_q_kernel<<<M_ * H_, 128, 0, stream>>>(q_f32, freqs, q16);
  kv_norm_rope_kernel<<<M_, 256, 0, stream>>>(kv_f32, freqs, kv_g, kv_b,
                                              kvlat16, krope16);

  // 4) kvb = kv_lat @ Wkv_b (fp32 out overlaps dead hs16/q_f32/kv_f32)
  gemm_bf16_nt<<<dim3((H_ * (NOPE_ + VD_)) / 128, M_ / 128), 256, 0, stream>>>(
      kvlat16, Wkvb_t, kvb_f32, H_ * (NOPE_ + VD_), KV_LORA_);

  // 5) split into K (nope||rope broadcast) and V, bf16
  split_kv_kernel<<<M_ * H_, 256, 0, stream>>>(kvb_f32, krope16, k16, v16);

  // 6) causal flash attention
  mla_attn_kernel<<<B_ * H_ * (T_ / 64), 128, 0, stream>>>(q16, k16, v16, attn16);

  // 7) out = attn @ Wout (fp32 result straight into d_out)
  gemm_bf16_nt<<<dim3(HID_ / 128, M_ / 128), 256, 0, stream>>>(
      attn16, Wout_t, out, HID_, H_ * VD_);
}
